// MoLoRALinear_9990093931086
// MI455X (gfx1250) — compile-verified, matrix operand order checked by reading, not run on hardware
//
#include <hip/hip_runtime.h>

// ---------------------------------------------------------------------------
// MoLoRA linear for MI455X (gfx1250, wave32, WMMA bf16 16x16x32, f32 accum)
//   out[T,O] = x@W^T + (diag(wmat) applied to (x@Acat^T)) @ Bflat^T
// Main GEMM: TDM (tensor_load_to_lds) double-buffered LDS pipeline.
// LoRA-h GEMM: VGPR-software-pipelined global->LDS staging.
// ---------------------------------------------------------------------------

#define T_TOK 8192   // B_*S_ = 4*2048
#define HDIM  2048
#define ODIM  2048
#define EEXP  8
#define RRANK 16
#define KLORA 128    // E*R
#define ALPHA 16.0f
#define LDSW  40     // LDS row stride in bf16 (80B = 64B row + 16B pad)

typedef __attribute__((ext_vector_type(16))) __bf16 v16bf;
typedef __attribute__((ext_vector_type(8)))  __bf16 v8bf;
typedef __attribute__((ext_vector_type(4)))  __bf16 v4bf;
typedef __attribute__((ext_vector_type(8)))  float  v8f;
typedef __attribute__((ext_vector_type(4)))  float  v4f;
typedef __attribute__((ext_vector_type(4)))  unsigned int u32x4;
typedef __attribute__((ext_vector_type(8)))  int    i32x8;
typedef __attribute__((ext_vector_type(4)))  int    i32x4;

#ifndef __has_builtin
#define __has_builtin(x) 0
#endif
#if __has_builtin(__builtin_amdgcn_tensor_load_to_lds) && \
    __has_builtin(__builtin_amdgcn_s_wait_tensorcnt)
#define USE_TDM 1
#else
#define USE_TDM 0
#endif

// ---------------------------------------------------------------------------
// fp32 -> bf16 bulk convert (n multiple of 4)
// ---------------------------------------------------------------------------
__global__ __launch_bounds__(256) void molora_cvt_bf16(
    const float* __restrict__ in, __bf16* __restrict__ out, int n) {
  int i = (blockIdx.x * 256 + threadIdx.x) * 4;
  if (i + 3 < n) {
    v4f v = *(const v4f*)(in + i);
    v4bf o;
    o.x = (__bf16)v.x; o.y = (__bf16)v.y;
    o.z = (__bf16)v.z; o.w = (__bf16)v.w;
    *(v4bf*)(out + i) = o;
  }
}

// ---------------------------------------------------------------------------
// Router: one wave32 per token -> dense wmat[T,E] (top-2 renormalized * alpha)
// ---------------------------------------------------------------------------
__global__ __launch_bounds__(32) void molora_router(
    const float* __restrict__ x, const float* __restrict__ gate_w,
    float* __restrict__ wmat) {
  const int t = blockIdx.x;
  const int lane = threadIdx.x;
  const float* xr = x + (size_t)t * HDIM;

  float acc[EEXP];
#pragma unroll
  for (int e = 0; e < EEXP; ++e) acc[e] = 0.f;

  for (int i = lane; i < HDIM; i += 32) {
    float xv = xr[i];
#pragma unroll
    for (int e = 0; e < EEXP; ++e)
      acc[e] = __builtin_fmaf(xv, gate_w[e * HDIM + i], acc[e]);
  }
#pragma unroll
  for (int e = 0; e < EEXP; ++e) {
#pragma unroll
    for (int off = 16; off > 0; off >>= 1)
      acc[e] += __shfl_xor(acc[e], off, 32);
  }
  float mx = acc[0];
#pragma unroll
  for (int e = 1; e < EEXP; ++e) mx = fmaxf(mx, acc[e]);
  float p[EEXP];
#pragma unroll
  for (int e = 0; e < EEXP; ++e) p[e] = __expf(acc[e] - mx);

  int i1 = 0; float b1 = p[0];
#pragma unroll
  for (int e = 1; e < EEXP; ++e) if (p[e] > b1) { b1 = p[e]; i1 = e; }
  int i2 = (i1 == 0) ? 1 : 0; float b2 = p[i2];
#pragma unroll
  for (int e = 0; e < EEXP; ++e)
    if (e != i1 && p[e] > b2) { b2 = p[e]; i2 = e; }

  float inv = 1.0f / (b1 + b2);
  if (lane < EEXP) {
    float w = (lane == i1) ? b1 * inv : (lane == i2) ? b2 * inv : 0.0f;
    wmat[(size_t)t * EEXP + lane] = w * ALPHA;
  }
}

// ---------------------------------------------------------------------------
// Repack B[E,O,R] (fp32) -> Bflat[O, E*R] (bf16), k = e*16 + r
// ---------------------------------------------------------------------------
__global__ __launch_bounds__(256) void molora_pack_b(
    const float* __restrict__ Bw, __bf16* __restrict__ bflat) {
  int idx = blockIdx.x * 256 + threadIdx.x;
  if (idx < ODIM * KLORA) {
    int o = idx >> 7, k = idx & 127, e = k >> 4, r = k & 15;
    bflat[idx] = (__bf16)Bw[((size_t)e * ODIM + o) * RRANK + r];
  }
}

// ---------------------------------------------------------------------------
// VGPR-pipelined tile staging helpers (128 rows x 32 k bf16, 16B chunks,
// 512 chunks / 256 threads = 2 per thread). fetch -> regs, store -> LDS.
// ---------------------------------------------------------------------------
struct TileRegs { v8bf v0, v1; };

__device__ __forceinline__ void fetch_tile(
    TileRegs& r, const __bf16* __restrict__ src, int ld, int rowBase, int k0,
    int tid) {
  int c0 = tid, c1 = tid + 256;
  const __bf16* g0 =
      src + (size_t)(rowBase + (c0 >> 2)) * ld + k0 + (c0 & 3) * 8;
  const __bf16* g1 =
      src + (size_t)(rowBase + (c1 >> 2)) * ld + k0 + (c1 & 3) * 8;
  r.v0 = *(const v8bf*)g0;
  r.v1 = *(const v8bf*)g1;
  __builtin_prefetch((const void*)(g0 + 32), 0, 1);
  __builtin_prefetch((const void*)(g1 + 32), 0, 1);
}

__device__ __forceinline__ void store_tile(__bf16* dst, const TileRegs& r,
                                           int tid) {
  int c0 = tid, c1 = tid + 256;
  *(v8bf*)(dst + (c0 >> 2) * LDSW + (c0 & 3) * 8) = r.v0;
  *(v8bf*)(dst + (c1 >> 2) * LDSW + (c1 & 3) * 8) = r.v1;
}

// 16x32 bf16 fragment per ISA lane layout:
// lane L (h=L>>4, r=L&15): row r, K = {h*8..h*8+7, 16+h*8..16+h*8+7}
__device__ __forceinline__ v16bf load_frag(
    const __bf16* __restrict__ tile, int rowBase, int lane) {
  int h = lane >> 4, r = lane & 15;
  const __bf16* p = tile + (rowBase + r) * LDSW + h * 8;
  v16bf f;
  *((v8bf*)&f)     = *(const v8bf*)p;
  *((v8bf*)&f + 1) = *(const v8bf*)(p + 16);
  return f;
}

// one 32-K step of the 32x64-per-wave MMA from staged LDS planes
__device__ __forceinline__ void mma_step(
    const __bf16* __restrict__ xsp, const __bf16* __restrict__ bsp,
    v8f (&acc)[2][4], int wm, int wn, int lane) {
  v16bf a0 = load_frag(xsp, wm * 32, lane);
  v16bf a1 = load_frag(xsp, wm * 32 + 16, lane);
#pragma unroll
  for (int ni = 0; ni < 4; ++ni) {
    v16bf b = load_frag(bsp, wn * 64 + ni * 16, lane);
    acc[0][ni] = __builtin_amdgcn_wmma_f32_16x16x32_bf16(
        false, a0, false, b, (short)0, acc[0][ni], false, false);
    acc[1][ni] = __builtin_amdgcn_wmma_f32_16x16x32_bf16(
        false, a1, false, b, (short)0, acc[1][ni], false, false);
  }
}

#if USE_TDM
// ---------------------------------------------------------------------------
// Tensor Data Mover: DMA a 128-row x 32-elem bf16 tile (64B rows) from a
// row-major [rows, ld] bf16 tensor at gptr (tile origin) into LDS at
// lds_byte_addr, inserting 16B of pad after every 64B -> LDS stride 80B.
// D# built per CDNA5 ISA 8.3/8.4 (2D: groups 2,3 zero).
// This toolchain exposes the 6-arg builtin form:
//   (u32x4 g0, i32x8 g1, i32x4 g2, i32x4 g3, i32x8 extra, i32 cpol)
// ---------------------------------------------------------------------------
__device__ __forceinline__ void tdm_load_tile(unsigned lds_byte_addr,
                                              const __bf16* gptr, int ld,
                                              int rows) {
  unsigned long long ga = (unsigned long long)(uintptr_t)gptr;
  u32x4 g0;
  g0[0] = 1u;                                   // count=1, user descriptor
  g0[1] = lds_byte_addr;                        // lds_addr
  g0[2] = (unsigned)(ga & 0xFFFFFFFFu);         // global_addr[31:0]
  g0[3] = (unsigned)((ga >> 32) & 0x01FFFFFFu)  // global_addr[56:32]
          | (2u << 30);                         // type=2 ("image")
  i32x8 g1;
  g1[0] = (1 << 16)     // data_size = 2 bytes
        | (1 << 20)     // pad_enable
        | (3 << 22)     // pad_interval: 16 DWORDs (64B) between pads
        | (3 << 25);    // pad_amount: 4 DWORDs (16B)
  g1[1] = (ld & 0xFFFF) << 16;                        // tensor_dim0[15:0]
  g1[2] = ((ld >> 16) & 0xFFFF) | ((rows & 0xFFFF) << 16);  // dim0 hi | dim1 lo
  g1[3] = ((rows >> 16) & 0xFFFF) | (32 << 16);       // dim1 hi | tile_dim0=32
  g1[4] = 128;                                        // tile_dim1=128, tile_dim2=0
  g1[5] = ld;                                         // tensor_dim0_stride[31:0]
  g1[6] = 0;                                          // stride hi | dim1_stride lo
  g1[7] = 0;
  i32x4 z4 = {0, 0, 0, 0};
  i32x8 z8 = {0, 0, 0, 0, 0, 0, 0, 0};
  __builtin_amdgcn_tensor_load_to_lds(g0, g1, z4, z4, z8, 0);
}
#endif

// ---------------------------------------------------------------------------
// LoRA-h GEMM: hw[T,128] = bf16( (x @ Acat^T) * wmat[t, n>>4] )
// Block tile 128(M) x 128(N=all), 8 waves of 32x64, VGPR-pipelined staging.
// ---------------------------------------------------------------------------
__global__ __launch_bounds__(256) void molora_h_gemm(
    const __bf16* __restrict__ xbf, const __bf16* __restrict__ abf,
    const float* __restrict__ wmat, __bf16* __restrict__ hwbf) {
  __shared__ __attribute__((aligned(16))) __bf16 xs[128 * LDSW];
  __shared__ __attribute__((aligned(16))) __bf16 bs[128 * LDSW];
  const int tid = threadIdx.x, lane = tid & 31, wid = tid >> 5;
  const int wm = wid & 3, wn = wid >> 2;
  const int tileM = blockIdx.x * 128;

  v8f acc[2][4] = {};

  TileRegs rx, rb;
  fetch_tile(rx, xbf, HDIM, tileM, 0, tid);
  fetch_tile(rb, abf, HDIM, 0, 0, tid);

  for (int k0 = 0; k0 < HDIM; k0 += 32) {
    store_tile(xs, rx, tid);
    store_tile(bs, rb, tid);
    __syncthreads();
    if (k0 + 32 < HDIM) {  // overlap next fetch with this step's MMAs
      fetch_tile(rx, xbf, HDIM, tileM, k0 + 32, tid);
      fetch_tile(rb, abf, HDIM, 0, k0 + 32, tid);
    }
    mma_step(xs, bs, acc, wm, wn, lane);
    __syncthreads();
  }

  // Each 16-wide N tile maps to exactly one expert: e = wn*4 + ni
  const int h = lane >> 4, r = lane & 15;
#pragma unroll
  for (int mi = 0; mi < 2; ++mi) {
#pragma unroll
    for (int ni = 0; ni < 4; ++ni) {
      int e = wn * 4 + ni;
      int n = wn * 64 + ni * 16 + r;
#pragma unroll
      for (int i = 0; i < 8; ++i) {
        int m = tileM + wm * 32 + mi * 16 + h * 8 + i;
        float v = acc[mi][ni][i] * wmat[(size_t)m * EEXP + e];
        hwbf[(size_t)m * KLORA + n] = (__bf16)v;
      }
    }
  }
}

// ---------------------------------------------------------------------------
// Main GEMM: out[T,O] = x@W^T (K=2048) + hw@Bflat^T (K=128, same accums).
// TDM double-buffered LDS pipeline (fallback: VGPR pipeline).
// ---------------------------------------------------------------------------
#define NSTEP0 (HDIM / 32)              // 64
#define NSTEPS (NSTEP0 + KLORA / 32)    // 68

__device__ __forceinline__ void main_step_src(
    int step, const __bf16* xbf, const __bf16* wbf, const __bf16* hwbf,
    const __bf16* bflat, int tileM, int tileN, const __bf16*& pa,
    const __bf16*& pb, int& ld, int& rowsA, int& rowsB) {
  if (step < NSTEP0) {
    int k0 = step * 32;
    ld = HDIM; rowsA = T_TOK; rowsB = ODIM;
    pa = xbf + (size_t)tileM * HDIM + k0;
    pb = wbf + (size_t)tileN * HDIM + k0;
  } else {
    int k0 = (step - NSTEP0) * 32;
    ld = KLORA; rowsA = T_TOK; rowsB = ODIM;
    pa = hwbf + (size_t)tileM * KLORA + k0;
    pb = bflat + (size_t)tileN * KLORA + k0;
  }
}

__global__ __launch_bounds__(256) void molora_main_gemm(
    const __bf16* __restrict__ xbf, const __bf16* __restrict__ wbf,
    const __bf16* __restrict__ hwbf, const __bf16* __restrict__ bflat,
    float* __restrict__ out) {
#if USE_TDM
  __shared__ __attribute__((aligned(16))) __bf16 xs[2][128 * LDSW];
  __shared__ __attribute__((aligned(16))) __bf16 bs[2][128 * LDSW];
#else
  __shared__ __attribute__((aligned(16))) __bf16 xs[1][128 * LDSW];
  __shared__ __attribute__((aligned(16))) __bf16 bs[1][128 * LDSW];
#endif
  const int tid = threadIdx.x, lane = tid & 31, wid = tid >> 5;
  const int wm = wid & 3, wn = wid >> 2;
  const int tileM = blockIdx.y * 128;
  const int tileN = blockIdx.x * 128;

  v8f acc[2][4] = {};
  const __bf16* pa; const __bf16* pb; int ld, rowsA, rowsB;

#if USE_TDM
  // prologue: DMA step 0 into plane 0
  if (wid == 0) {
    main_step_src(0, xbf, wbf, hwbf, bflat, tileM, tileN, pa, pb, ld, rowsA,
                  rowsB);
    tdm_load_tile((unsigned)(size_t)(const void*)&xs[0][0], pa, ld, rowsA);
    tdm_load_tile((unsigned)(size_t)(const void*)&bs[0][0], pb, ld, rowsB);
    __builtin_amdgcn_s_wait_tensorcnt((short)0);
  }
  __syncthreads();

  int cur = 0;
#pragma unroll 1
  for (int step = 0; step < NSTEPS; ++step) {
    if (wid == 0 && step + 1 < NSTEPS) {  // DMA next tile into other plane
      main_step_src(step + 1, xbf, wbf, hwbf, bflat, tileM, tileN, pa, pb, ld,
                    rowsA, rowsB);
      tdm_load_tile((unsigned)(size_t)(const void*)&xs[cur ^ 1][0], pa, ld,
                    rowsA);
      tdm_load_tile((unsigned)(size_t)(const void*)&bs[cur ^ 1][0], pb, ld,
                    rowsB);
    }
    mma_step(&xs[cur][0], &bs[cur][0], acc, wm, wn, lane);  // overlaps DMA
    if (wid == 0 && step + 1 < NSTEPS)
      __builtin_amdgcn_s_wait_tensorcnt((short)0);
    __syncthreads();
    cur ^= 1;
  }
#else
  // fallback: VGPR-software-pipelined staging (also used by host pass)
  TileRegs rx, rb;
  main_step_src(0, xbf, wbf, hwbf, bflat, tileM, tileN, pa, pb, ld, rowsA,
                rowsB);
  fetch_tile(rx, pa, ld, 0, 0, tid);
  fetch_tile(rb, pb, ld, 0, 0, tid);
#pragma unroll 1
  for (int step = 0; step < NSTEPS; ++step) {
    store_tile(&xs[0][0], rx, tid);
    store_tile(&bs[0][0], rb, tid);
    __syncthreads();
    if (step + 1 < NSTEPS) {
      main_step_src(step + 1, xbf, wbf, hwbf, bflat, tileM, tileN, pa, pb, ld,
                    rowsA, rowsB);
      fetch_tile(rx, pa, ld, 0, 0, tid);
      fetch_tile(rb, pb, ld, 0, 0, tid);
    }
    mma_step(&xs[0][0], &bs[0][0], acc, wm, wn, lane);
    __syncthreads();
  }
#endif

  const int h = lane >> 4, r = lane & 15;
#pragma unroll
  for (int mi = 0; mi < 2; ++mi) {
#pragma unroll
    for (int ni = 0; ni < 4; ++ni) {
      int n = tileN + wn * 64 + ni * 16 + r;
#pragma unroll
      for (int i = 0; i < 8; ++i) {
        int m = tileM + wm * 32 + mi * 16 + h * 8 + i;
        out[(size_t)m * ODIM + n] = acc[mi][ni][i];
      }
    }
  }
}

// ---------------------------------------------------------------------------
// Launch
// ---------------------------------------------------------------------------
extern "C" void kernel_launch(void* const* d_in, const int* in_sizes, int n_in,
                              void* d_out, int out_size, void* d_ws, size_t ws_size,
                              hipStream_t stream) {
  const float* x      = (const float*)d_in[0];  // [4,2048,2048]
  const float* W      = (const float*)d_in[1];  // [2048,2048]
  const float* gate_w = (const float*)d_in[2];  // [8,2048]
  const float* A      = (const float*)d_in[3];  // [8,16,2048] == Acat[128,2048]
  const float* Bw     = (const float*)d_in[4];  // [8,2048,16]
  float* out = (float*)d_out;                   // [8192,2048]

  char* ws = (char*)d_ws;
  float*  wmat  = (float*)(ws);                      // 8192*8*4    = 256KB
  __bf16* xbf   = (__bf16*)(ws + (size_t)262144);    // 8192*2048*2 = 32MB
  __bf16* wbf   = (__bf16*)(ws + (size_t)33816576);  // 2048*2048*2 = 8MB
  __bf16* abf   = (__bf16*)(ws + (size_t)42205184);  // 128*2048*2  = 512KB
  __bf16* hwbf  = (__bf16*)(ws + (size_t)42729472);  // 8192*128*2  = 2MB
  __bf16* bflat = (__bf16*)(ws + (size_t)44826624);  // 2048*128*2  = 512KB

  molora_cvt_bf16<<<(T_TOK * HDIM / 4 + 255) / 256, 256, 0, stream>>>(
      x, xbf, T_TOK * HDIM);
  molora_cvt_bf16<<<(ODIM * HDIM / 4 + 255) / 256, 256, 0, stream>>>(
      W, wbf, ODIM * HDIM);
  molora_cvt_bf16<<<(KLORA * HDIM / 4 + 255) / 256, 256, 0, stream>>>(
      A, abf, KLORA * HDIM);

  molora_router<<<T_TOK, 32, 0, stream>>>(x, gate_w, wmat);

  molora_pack_b<<<(ODIM * KLORA + 255) / 256, 256, 0, stream>>>(Bw, bflat);

  molora_h_gemm<<<T_TOK / 128, 256, 0, stream>>>(xbf, abf, wmat, hwbf);

  molora_main_gemm<<<dim3(ODIM / 128, T_TOK / 128), 256, 0, stream>>>(
      xbf, wbf, hwbf, bflat, out);
}
